// Dynamic_Actor_19739669693136
// MI455X (gfx1250) — matrix operand order repaired and checked by reading, not compiled, not run on hardware
//
#include <hip/hip_runtime.h>
#include <hip/hip_bf16.h>

// ---------------- CDNA5 WMMA plumbing ----------------
typedef __attribute__((ext_vector_type(16))) __bf16 v16bf;
typedef __attribute__((ext_vector_type(8)))  float  v8f;

#define WMMA_BF16(a, b, c) \
  __builtin_amdgcn_wmma_f32_16x16x32_bf16(false, (a), false, (b), (short)0, (c), false, false)

// A-matrix (16-bit, 16x32) per-lane K mapping (ISA 7.12.2):
// lanes 0-15: VGPR0..3 -> K0..7, VGPR4..7 -> K16..23 ; lanes 16-31: +8
__device__ __forceinline__ int kmap(int e, int hi) {
  return ((e & 8) << 1) + hi * 8 + (e & 7);
}

// ---------------- CDNA5 async global->LDS (ASYNCcnt path) ----------------
__device__ __forceinline__ void async_load_b128(unsigned lds_off, const void* gaddr) {
  asm volatile("global_load_async_to_lds_b128 %0, %1, off"
               :: "v"(lds_off), "v"(gaddr) : "memory");
}
__device__ __forceinline__ void wait_async0() {
  asm volatile("s_wait_asynccnt 0" ::: "memory");
}

constexpr int kB   = 256;
constexpr int kT   = 256;
constexpr int kOBS = 512;
constexpr int kH   = 64;
constexpr int kNT  = kB * kT;          // 65536 rows

__device__ __forceinline__ float sigm(float x) { return 1.f / (1.f + __expf(-x)); }

// ============================================================
// Kernel 1: x1 = relu(LN(ud @ dense_w + dense_b)) -> bf16
// grid 512, block 256 (8 waves), 128 rows/block, N=64, K=512
// ============================================================
__global__ __launch_bounds__(256)
void dense_ln_kernel(const float* __restrict__ ud, const float* __restrict__ W,
                     const float* __restrict__ bias, const float* __restrict__ g,
                     const float* __restrict__ bvec, __bf16* __restrict__ x1)
{
  __shared__ __bf16 lds_a[128 * 32];
  __shared__ __bf16 lds_b[32 * 64];
  __shared__ float  lds_c[128 * 64];
  const int tid  = threadIdx.x;
  const int wave = tid >> 5;
  const int lane = tid & 31;
  const int row0 = blockIdx.x * 128;
  const int hi   = lane >> 4;
  const int arow = wave * 16 + (lane & 15);

  v8f acc[4];
#pragma unroll
  for (int j = 0; j < 4; j++)
#pragma unroll
    for (int r = 0; r < 8; r++) acc[j][r] = 0.f;

  for (int kk = 0; kk < kOBS; kk += 32) {
    for (int i = tid; i < 128 * 32; i += 256) {
      int r = i >> 5, c = i & 31;
      lds_a[i] = (__bf16)ud[(size_t)(row0 + r) * kOBS + kk + c];
    }
    for (int i = tid; i < 32 * 64; i += 256) {
      int r = i >> 6, c = i & 63;
      lds_b[i] = (__bf16)W[(size_t)(kk + r) * kH + c];
    }
    __syncthreads();
    v16bf afrag;
#pragma unroll
    for (int e = 0; e < 16; e++) afrag[e] = lds_a[arow * 32 + kmap(e, hi)];
#pragma unroll
    for (int j = 0; j < 4; j++) {
      v16bf bfrag;
#pragma unroll
      for (int e = 0; e < 16; e++) bfrag[e] = lds_b[kmap(e, hi) * 64 + j * 16 + (lane & 15)];
      acc[j] = WMMA_BF16(afrag, bfrag, acc[j]);
    }
    __syncthreads();
  }
  // dump C tiles (+bias) to LDS for row-wise LayerNorm
#pragma unroll
  for (int j = 0; j < 4; j++)
#pragma unroll
    for (int r = 0; r < 8; r++) {
      int m = r + hi * 8;
      int n = j * 16 + (lane & 15);
      lds_c[(wave * 16 + m) * 64 + n] = acc[j][r] + bias[n];
    }
  __syncthreads();
  if (tid < 128) {
    float s = 0.f, s2 = 0.f;
#pragma unroll 8
    for (int n = 0; n < 64; n++) { float v = lds_c[tid * 64 + n]; s += v; s2 += v * v; }
    float mu = s * (1.f / 64.f);
    float var = s2 * (1.f / 64.f) - mu * mu;
    float rs = rsqrtf(var + 1e-3f);
    size_t base = (size_t)(row0 + tid) * kH;
#pragma unroll 8
    for (int n = 0; n < 64; n++) {
      float v = (lds_c[tid * 64 + n] - mu) * rs * g[n] + bvec[n];
      x1[base + n] = (__bf16)(v > 0.f ? v : 0.f);
    }
  }
}

// ============================================================
// Kernel 2: xw[dir][t][b] = x1[b*T+t] @ {fw_k|bw_k} + bias
// (M=NT,K=64,N=256). grid 8192 (dir = bid&1, 16 rows/block),
// block 256 (8 waves, 32 cols each). Output layout is t-major so the
// scan kernel reads one contiguous 16KB tile per timestep.
// ============================================================
__global__ __launch_bounds__(256)
void xw_kernel(const __bf16* __restrict__ x1,
               const float* __restrict__ fw_k, const float* __restrict__ fw_b,
               const float* __restrict__ bw_k, const float* __restrict__ bw_b,
               float* __restrict__ xw)        // [2][T][B][256]
{
  __shared__ __bf16 lds_x[16 * 64];
  __shared__ __bf16 lds_w[64 * 256];
  const int tid = threadIdx.x, wave = tid >> 5, lane = tid & 31;
  const int dir = blockIdx.x & 1;
  const int row0 = (blockIdx.x >> 1) * 16;     // rows row0..row0+15 share batch b
  const int brow = row0 >> 8;                  // batch index
  const int t0   = row0 & 255;                 // first timestep in tile
  const float* Wk   = dir ? bw_k : fw_k;
  const float* bias = dir ? bw_b : fw_b;

  for (int i = tid; i < 16 * 64; i += 256) lds_x[i] = x1[(size_t)row0 * kH + i];
  for (int i = tid; i < 64 * 256; i += 256) lds_w[i] = (__bf16)Wk[i];
  __syncthreads();

  const int n0 = wave * 32;
  const int hi = lane >> 4, arow = lane & 15;
  v8f acc[2];
#pragma unroll
  for (int j = 0; j < 2; j++)
#pragma unroll
    for (int r = 0; r < 8; r++) acc[j][r] = 0.f;

#pragma unroll
  for (int ks = 0; ks < 2; ks++) {
    v16bf afrag;
#pragma unroll
    for (int e = 0; e < 16; e++) afrag[e] = lds_x[arow * 64 + ks * 32 + kmap(e, hi)];
#pragma unroll
    for (int j = 0; j < 2; j++) {
      v16bf bfrag;
#pragma unroll
      for (int e = 0; e < 16; e++) bfrag[e] = lds_w[(ks * 32 + kmap(e, hi)) * 256 + n0 + j * 16 + (lane & 15)];
      acc[j] = WMMA_BF16(afrag, bfrag, acc[j]);
    }
  }
  size_t obase = (size_t)dir * kNT * 256;
#pragma unroll
  for (int j = 0; j < 2; j++)
#pragma unroll
    for (int r = 0; r < 8; r++) {
      int m = r + hi * 8;                      // timestep offset within tile
      int n = n0 + j * 16 + (lane & 15);
      xw[obase + (size_t)((t0 + m) * kB + brow) * 256 + n] = acc[j][r] + bias[n];
    }
}

// ============================================================
// Kernel 3: masked LSTM scan, both directions.
// grid 32 (dir = bid>>4, 16 batches/block), block 512 = 16 waves.
// Per step: z = xw[t] + h @ Wr  (M=16,K=64,N=256) -> gates -> h,c.
// out == h (mask t<au is monotone), so hbuf[t] = h after step t.
// Wr B-fragments live in registers; the xw tile for step s+1 is
// double-buffered in LDS via global_load_async_to_lds_b128.
// ============================================================
__global__ __launch_bounds__(512)
void lstm_scan_kernel(const float* __restrict__ xw,     // [2][T][B][256]
                      const float* __restrict__ fw_r, const float* __restrict__ bw_r,
                      const int* __restrict__ au,
                      float* __restrict__ hbuf)          // [2][NT][64]
{
  __shared__ __bf16 lds_h[16 * 64];
  __shared__ float  lds_z[16 * 256];
  __shared__ float  lds_xw[2][16 * 256];
  const int tid = threadIdx.x, wave = tid >> 5, lane = tid & 31;
  const int dir = blockIdx.x >> 4;
  const int b0  = (blockIdx.x & 15) * 16;
  const float* Wr = dir ? bw_r : fw_r;

  const int n0 = wave * 16;
  const int hi = lane >> 4, arow = lane & 15;
  const size_t xbase = (size_t)dir * kNT * 256;

  // loop-invariant B fragments of Wr: 2 k-steps x (16 cols per wave)
  v16bf wrf[2];
#pragma unroll
  for (int ks = 0; ks < 2; ks++)
#pragma unroll
    for (int e = 0; e < 16; e++)
      wrf[ks][e] = (__bf16)Wr[(size_t)(ks * 32 + kmap(e, hi)) * 256 + n0 + (lane & 15)];

  for (int i = tid; i < 16 * 64; i += 512) lds_h[i] = (__bf16)0.f;

  // gate-phase ownership: thread -> (row gm, cols gc, gc+1)
  const int gm = tid >> 5;
  const int gc = (tid & 31) * 2;
  const int alen = au[b0 + gm];
  float c0 = 0.f, c1 = 0.f, h0v = 0.f, h1v = 0.f;

  // prologue: async-load the first xw tile (16 batches x 256) into buf 0
  {
    const int tfirst = dir ? (kT - 1) : 0;
    const char* gsrc = (const char*)(xw + xbase + (size_t)(tfirst * kB + b0) * 256);
    unsigned lbase = (unsigned)(uintptr_t)(&lds_xw[0][0]);
#pragma unroll
    for (int q = 0; q < 2; q++) {
      unsigned off = (unsigned)(q * 512 + tid) * 16u;
      async_load_b128(lbase + off, gsrc + off);
    }
    wait_async0();
  }
  __syncthreads();

  for (int step = 0; step < kT; step++) {
    const int t   = dir ? (kT - 1 - step) : step;
    const int buf = step & 1;
    // acc <- xw tile (LDS, prefetched asynchronously)
    v8f acc;
#pragma unroll
    for (int r = 0; r < 8; r++)
      acc[r] = lds_xw[buf][(r + hi * 8) * 256 + n0 + (lane & 15)];
    // z += h @ Wr
#pragma unroll
    for (int ks = 0; ks < 2; ks++) {
      v16bf afrag;
#pragma unroll
      for (int e = 0; e < 16; e++) afrag[e] = lds_h[arow * 64 + ks * 32 + kmap(e, hi)];
      acc = WMMA_BF16(afrag, wrf[ks], acc);
    }
#pragma unroll
    for (int r = 0; r < 8; r++)
      lds_z[(r + hi * 8) * 256 + n0 + (lane & 15)] = acc[r];
    __syncthreads();
    // gates (diverges only on masked lanes; no WMMA inside)
    if (t < alen) {
      const float* z = &lds_z[gm * 256];
      float ii = sigm(z[gc]),        ff = sigm(z[64 + gc]);
      float gg = tanhf(z[128 + gc]), oo = sigm(z[192 + gc]);
      c0 = ff * c0 + ii * gg;
      h0v = oo * tanhf(c0);
      ii = sigm(z[gc + 1]);          ff = sigm(z[64 + gc + 1]);
      gg = tanhf(z[128 + gc + 1]);   oo = sigm(z[192 + gc + 1]);
      c1 = ff * c1 + ii * gg;
      h1v = oo * tanhf(c1);
    }
    lds_h[gm * 64 + gc]     = (__bf16)h0v;
    lds_h[gm * 64 + gc + 1] = (__bf16)h1v;
    size_t ob = (size_t)dir * kNT * kH + (size_t)((b0 + gm) * kT + t) * kH;
    hbuf[ob + gc]     = h0v;
    hbuf[ob + gc + 1] = h1v;
    // issue async copy of next step's tile into the other buffer
    if (step + 1 < kT) {
      const int tn = dir ? (t - 1) : (t + 1);
      const char* gsrc = (const char*)(xw + xbase + (size_t)(tn * kB + b0) * 256);
      unsigned lbase = (unsigned)(uintptr_t)(&lds_xw[buf ^ 1][0]);
#pragma unroll
      for (int q = 0; q < 2; q++) {
        unsigned off = (unsigned)(q * 512 + tid) * 16u;
        async_load_b128(lbase + off, gsrc + off);
      }
    }
    wait_async0();
    __syncthreads();
  }
}

// ============================================================
// Kernel 4: out = tanh(relu(LN(concat(fwd,bwd))) @ conv_w + conv_b)
// grid 512 (128 rows/block), block 256 (8 waves), K=128, N=64
// ============================================================
__global__ __launch_bounds__(256)
void final_kernel(const float* __restrict__ hbuf,   // [2][NT][64]
                  const float* __restrict__ ln_g, const float* __restrict__ ln_b,
                  const float* __restrict__ conv_w, const float* __restrict__ conv_b,
                  float* __restrict__ out)           // [NT][64]
{
  __shared__ __bf16 lds_x[128 * 128];
  __shared__ __bf16 lds_w[128 * 64];
  const int tid = threadIdx.x, wave = tid >> 5, lane = tid & 31;
  const int row0 = blockIdx.x * 128;

  for (int i = tid; i < 128 * 64; i += 256) lds_w[i] = (__bf16)conv_w[i];

  // LN over 128 features: 2 threads per row (half 0 = fwd, half 1 = bwd)
  {
    const int row = tid >> 1, half = tid & 1;
    const float* src = hbuf + (size_t)half * kNT * kH + (size_t)(row0 + row) * kH;
    float s = 0.f, s2 = 0.f;
#pragma unroll 8
    for (int n = 0; n < 64; n++) { float v = src[n]; s += v; s2 += v * v; }
    s  += __shfl_xor(s, 1, 32);
    s2 += __shfl_xor(s2, 1, 32);
    float mu = s * (1.f / 128.f);
    float var = s2 * (1.f / 128.f) - mu * mu;
    float rs = rsqrtf(var + 1e-3f);
#pragma unroll 8
    for (int n = 0; n < 64; n++) {
      int f = half * 64 + n;
      float v = (src[n] - mu) * rs * ln_g[f] + ln_b[f];
      lds_x[row * 128 + f] = (__bf16)(v > 0.f ? v : 0.f);
    }
  }
  __syncthreads();

  const int hi = lane >> 4, arow = lane & 15;
  v8f acc[4];
#pragma unroll
  for (int j = 0; j < 4; j++) {
    float bj = conv_b[j * 16 + (lane & 15)];
#pragma unroll
    for (int r = 0; r < 8; r++) acc[j][r] = bj;
  }
#pragma unroll
  for (int ks = 0; ks < 4; ks++) {
    v16bf afrag;
#pragma unroll
    for (int e = 0; e < 16; e++)
      afrag[e] = lds_x[(wave * 16 + arow) * 128 + ks * 32 + kmap(e, hi)];
#pragma unroll
    for (int j = 0; j < 4; j++) {
      v16bf bfrag;
#pragma unroll
      for (int e = 0; e < 16; e++) bfrag[e] = lds_w[(ks * 32 + kmap(e, hi)) * 64 + j * 16 + (lane & 15)];
      acc[j] = WMMA_BF16(afrag, bfrag, acc[j]);
    }
  }
#pragma unroll
  for (int j = 0; j < 4; j++)
#pragma unroll
    for (int r = 0; r < 8; r++) {
      int m = r + hi * 8;
      out[(size_t)(row0 + wave * 16 + m) * kH + j * 16 + (lane & 15)] = tanhf(acc[j][r]);
    }
}

// ============================================================
extern "C" void kernel_launch(void* const* d_in, const int* in_sizes, int n_in,
                              void* d_out, int out_size, void* d_ws, size_t ws_size,
                              hipStream_t stream)
{
  const float* ud      = (const float*)d_in[0];
  // d_in[1] = mask (unused by reference math; seq_mask comes from au)
  const int*   au      = (const int*)d_in[2];
  const float* dense_w = (const float*)d_in[3];
  const float* dense_b = (const float*)d_in[4];
  const float* ln1_g   = (const float*)d_in[5];
  const float* ln1_b   = (const float*)d_in[6];
  const float* fw_k    = (const float*)d_in[7];
  const float* fw_r    = (const float*)d_in[8];
  const float* fw_b    = (const float*)d_in[9];
  const float* bw_k    = (const float*)d_in[10];
  const float* bw_r    = (const float*)d_in[11];
  const float* bw_b    = (const float*)d_in[12];
  const float* ln3_g   = (const float*)d_in[13];
  const float* ln3_b   = (const float*)d_in[14];
  const float* conv_w  = (const float*)d_in[15];
  const float* conv_b  = (const float*)d_in[16];
  float* out = (float*)d_out;

  char* ws = (char*)d_ws;
  __bf16* x1  = (__bf16*)ws;                                   // NT*64 bf16 = 8 MB
  float*  xw  = (float*)(ws + (size_t)kNT * kH * 2);           // 2*NT*256 f32 = 128 MB
  float*  hb  = (float*)(ws + (size_t)kNT * kH * 2
                            + (size_t)2 * kNT * 256 * 4);      // 2*NT*64 f32 = 32 MB

  dense_ln_kernel<<<kNT / 128, 256, 0, stream>>>(ud, dense_w, dense_b, ln1_g, ln1_b, x1);
  xw_kernel<<<(kNT / 16) * 2, 256, 0, stream>>>(x1, fw_k, fw_b, bw_k, bw_b, xw);
  lstm_scan_kernel<<<32, 512, 0, stream>>>(xw, fw_r, bw_r, au, hb);
  final_kernel<<<kNT / 128, 256, 0, stream>>>(hb, ln3_g, ln3_b, conv_w, conv_b, out);
}
